// SubtaskHistoryEncoder_6347961663761
// MI455X (gfx1250) — matrix operand order repaired and checked by reading, not compiled, not run on hardware
//
#include <hip/hip_runtime.h>

// ---------------------------------------------------------------------------
// SubtaskHistoryEncoder for MI455X (gfx1250): bf16 WMMA + TDM-staged GEMM.
// D=512, H=8, dh=64, FF=512, L=3072 tokens (1 SOS + 3071 history).
// ---------------------------------------------------------------------------

#define T_TYPES 15
#define N_ARGS  73
#define N_MAPS  3
#define DMODEL  512
#define NHEAD   8
#define DHEAD   64
#define FFDIM   512
#define NTOK    3071
#define LSEQ    3072
#define LN_EPS  1e-5f
#define LRELU   0.01f

typedef __bf16 bf16_t;
typedef __bf16 v16bf __attribute__((ext_vector_type(16)));
typedef __bf16 v8bf  __attribute__((ext_vector_type(8)));
typedef float  v8f   __attribute__((ext_vector_type(8)));
typedef unsigned int u32x4 __attribute__((ext_vector_type(4)));
typedef int i32x8 __attribute__((ext_vector_type(8)));
typedef int i32x4 __attribute__((ext_vector_type(4)));

// ---------------------------------------------------------------------------
// TDM: issue a 2D tensor tile load (tile_dim0=32 cols x tile_dim1=64 rows,
// 2-byte elements, row stride = K elements) from global into LDS.
// D# bitfields per CDNA5 ISA ch.8 (group0 128b, group1 256b).
// This toolchain's builtin takes 6 args (g0, g1, g2, g3, g4, cpol).
// ---------------------------------------------------------------------------
__device__ __forceinline__ void tdm_load_tile_64x32(unsigned long long ga,
                                                    unsigned lds_addr,
                                                    int K, int Ntot) {
  u32x4 g0;
  g0[0] = 1u;                                            // count=1 (valid user D#)
  g0[1] = lds_addr;                                      // lds_addr (bytes)
  g0[2] = (unsigned)(ga & 0xFFFFFFFFull);                // global_addr[31:0]
  g0[3] = (unsigned)((ga >> 32) & 0x01FFFFFFull)         // global_addr[56:32]
        | 0x80000000u;                                   // type=2 ("image")
  i32x8 g1;
  g1[0] = 0x00010000;                                    // data_size=1 (2 bytes)
  g1[1] = (K & 0xFFFF) << 16;                            // tensor_dim0[15:0]
  g1[2] = ((K >> 16) & 0xFFFF) | ((Ntot & 0xFFFF) << 16);// dim0[31:16] | dim1[15:0]
  g1[3] = ((Ntot >> 16) & 0xFFFF) | (32 << 16);          // dim1[31:16] | tile_dim0=32
  g1[4] = 64;                                            // tile_dim1=64 (tile_dim2=0)
  g1[5] = K;                                             // tensor_dim0_stride[31:0]
  g1[6] = 0;                                             // stride[47:32] | dim1_stride lo
  g1[7] = 0;
  i32x4 z4 = {0, 0, 0, 0};
  i32x8 z8 = {0, 0, 0, 0, 0, 0, 0, 0};
  __builtin_amdgcn_tensor_load_to_lds(g0, g1, z4, z4, z8, 0);
}

// ---------------------------------------------------------------------------
// fp32 -> bf16 conversion (weights)
// ---------------------------------------------------------------------------
__global__ void f32_to_bf16_kernel(const float* __restrict__ in,
                                   bf16_t* __restrict__ out, int n) {
  for (int i = blockIdx.x * blockDim.x + threadIdx.x; i < n;
       i += gridDim.x * blockDim.x)
    out[i] = (bf16_t)in[i];
}

// ---------------------------------------------------------------------------
// bid[i]: segment id.  bid[0] = -1 (SOS), bid[1+i] = cumsum(seq_masks==0).
// ---------------------------------------------------------------------------
__global__ void bid_kernel(const int* __restrict__ sm, int* __restrict__ bid) {
  if (threadIdx.x == 0) {
    bid[0] = -1;
    int run = 0;
    for (int i = 0; i < NTOK; ++i) {
      run += (sm[i] == 0) ? 1 : 0;
      bid[i + 1] = run;
    }
  }
}

// ---------------------------------------------------------------------------
// Embedding gather: x[tok] = Wt.T[t]+bt + Wa.T[a]+ba + Wm.T[m]+bm  (bf16 out)
// ---------------------------------------------------------------------------
__global__ void embed_kernel(const int* __restrict__ hist,
                             const float* __restrict__ Wt, const float* __restrict__ bt,
                             const float* __restrict__ Wa, const float* __restrict__ ba,
                             const float* __restrict__ Wm, const float* __restrict__ bm,
                             const float* __restrict__ sos,
                             bf16_t* __restrict__ xb) {
  const int tok = blockIdx.x;
  const int tid = threadIdx.x;            // 256 threads, 2 elems each
  if (tok == 0) {
    xb[tid]       = (bf16_t)sos[tid];
    xb[tid + 256] = (bf16_t)sos[tid + 256];
    return;
  }
  const int idx = hist[tok - 1];
  const int t = idx / (N_ARGS * N_MAPS);
  const int a = (idx / N_MAPS) % N_ARGS;
  const int m = idx % N_MAPS;
#pragma unroll
  for (int p = 0; p < 2; ++p) {
    const int d = tid + p * 256;
    float v = Wt[d * T_TYPES + t] + bt[d] + Wa[d * N_ARGS + a] + ba[d] +
              Wm[d * N_MAPS + m] + bm[d];
    xb[(size_t)tok * DMODEL + d] = (bf16_t)v;
  }
}

// ---------------------------------------------------------------------------
// bf16 GEMM: C[M,N] = (A[M,K] @ W[N,K]^T + bias) * scale,  opt leaky-ReLU.
// Block = 128 threads (4 waves); all 4 waves share the B tile (cols
// nBase..nBase+63), which the Tensor Data Mover double-buffers into LDS.
// Per k-step: 1 TDM prefetch (wave 0), 1 workgroup barrier, batched ds_load
// of 4 B fragments (distinct regs -> staged s_wait_dscnt), global-load next
// A fragment, 4 WMMAs per wave.  Layouts per CDNA5 ISA 7.12.2 (wave32).
// ---------------------------------------------------------------------------
__global__ __launch_bounds__(128)
void gemm_bf16_kernel(const bf16_t* __restrict__ A, const bf16_t* __restrict__ W,
                      const float* __restrict__ bias,
                      float* __restrict__ outF, bf16_t* __restrict__ outB,
                      bf16_t* __restrict__ outT,
                      int M, int N, int K, int act, float scale) {
  __shared__ __attribute__((aligned(64))) bf16_t Btile[2][64 * 32];

  const int wave = threadIdx.x >> 5;
  const int lane = threadIdx.x & 31;
  const int lh = lane >> 4;
  const int lr = lane & 15;

  const int mTile = blockIdx.x * 64 + wave * 16;
  const int nBase = blockIdx.y * 64;
  const int nk = K >> 5;

  const unsigned long long gaBase =
      (unsigned long long)(uintptr_t)(W + (size_t)nBase * K);
  const unsigned ldsOff0 = (unsigned)(uintptr_t)(&Btile[0][0]);
  const unsigned ldsOff1 = (unsigned)(uintptr_t)(&Btile[1][0]);

  if (wave == 0)   // stage tile 0
    tdm_load_tile_64x32(gaBase, ldsOff0, K, N);

  v8f acc[4] = {};
  const bf16_t* aRow = A + (size_t)(mTile + lr) * K + lh * 16;
  v8bf a0c = *(const v8bf*)(aRow);
  v8bf a1c = *(const v8bf*)(aRow + 8);

  for (int kk = 0; kk < nk; ++kk) {
    if (wave == 0)
      __builtin_amdgcn_s_wait_tensorcnt((short)0);   // tile kk resident
    __syncthreads();                                 // publish tile kk; prev reads done
    if (wave == 0 && kk + 1 < nk)                    // prefetch tile kk+1
      tdm_load_tile_64x32(gaBase + (unsigned long long)(kk + 1) * 64,
                          ((kk + 1) & 1) ? ldsOff1 : ldsOff0, K, N);

    // prefetch next A fragment (global) while WMMAs run
    v8bf a0n = {}, a1n = {};
    if (kk + 1 < nk) {
      a0n = *(const v8bf*)(aRow + (kk + 1) * 32);
      a1n = *(const v8bf*)(aRow + (kk + 1) * 32 + 8);
    }

    v16bf af;
#pragma unroll
    for (int i = 0; i < 8; ++i) { af[i] = a0c[i]; af[i + 8] = a1c[i]; }

    // batch all 4 B-fragment loads into distinct registers, then compute
    const bf16_t* bt = &Btile[kk & 1][0];
    v16bf bfW[4];
#pragma unroll
    for (int t = 0; t < 4; ++t)
      bfW[t] = *(const v16bf*)(bt + (t * 16 + lr) * 32 + lh * 16);
#pragma unroll
    for (int t = 0; t < 4; ++t)
      acc[t] = __builtin_amdgcn_wmma_f32_16x16x32_bf16(
          false, af, false, bfW[t], (short)0, acc[t], false, false);

    a0c = a0n; a1c = a1n;
  }

#pragma unroll
  for (int t = 0; t < 4; ++t) {
    const int n = nBase + t * 16 + lr;
    const float b = bias ? bias[n] : 0.f;
#pragma unroll
    for (int r = 0; r < 8; ++r) {
      const int m = mTile + lh * 8 + r;
      float v = (acc[t][r] + b) * scale;
      if (act) v = (v > 0.f) ? v : LRELU * v;
      if (outF) outF[(size_t)m * N + n] = v;
      if (outB) outB[(size_t)m * N + n] = (bf16_t)v;
      if (outT) outT[(size_t)n * M + m] = (bf16_t)v;   // V^T store: [N][M]
    }
  }
}

// ---------------------------------------------------------------------------
// Flash attention (one wave per (head, 16-row block)), 64 keys per step.
//   Q,K: [L, D] bf16 row-major (Q pre-scaled by 1/8).  Vt: [D][L] bf16.
//   mask allows j==0 or (bid match && j<i).  S is 16x64 (4 fragments),
//   P -> two 16x32 A fragments via LDS transpose, PV = 8 WMMAs.
//   Max key index < ceil((iBlk+1)/4)*64 <= 3072: always in range.
// ---------------------------------------------------------------------------
__global__ __launch_bounds__(32)
void attn_kernel(const bf16_t* __restrict__ Q, const bf16_t* __restrict__ Kb,
                 const bf16_t* __restrict__ Vt, const int* __restrict__ bid,
                 bf16_t* __restrict__ O) {
  const int iBlk = blockIdx.x;            // 0..191
  const int h    = blockIdx.y;            // 0..7
  const int lane = threadIdx.x & 31;
  const int lh = lane >> 4;
  const int lr = lane & 15;

  __shared__ __attribute__((aligned(64))) bf16_t Plds[16 * 64];

  // Q fragments for k = 0..31 and 32..63 of this head
  v16bf qf[2];
  const bf16_t* qRow = Q + (size_t)(iBlk * 16 + lr) * DMODEL + h * DHEAD + lh * 16;
#pragma unroll
  for (int t = 0; t < 2; ++t) {
    v8bf a0 = *(const v8bf*)(qRow + t * 32);
    v8bf a1 = *(const v8bf*)(qRow + t * 32 + 8);
#pragma unroll
    for (int i = 0; i < 8; ++i) { qf[t][i] = a0[i]; qf[t][i + 8] = a1[i]; }
  }

  float rowM[8], rowS[8];
  int bi[8];
#pragma unroll
  for (int r = 0; r < 8; ++r) {
    rowM[r] = -3.0e38f; rowS[r] = 0.f;
    bi[r] = bid[iBlk * 16 + lh * 8 + r];
  }
  v8f oAcc[4] = {};

  const int nQuads = (iBlk + 4) >> 2;     // ceil((iBlk+1)/4)
  for (int jq = 0; jq < nQuads; ++jq) {
    const int jBase = jq * 64;

    // S = Q @ K^T for 64 keys (four 16x16 fragments); batch loads per k-step
    v8f s[4] = {};
#pragma unroll
    for (int t = 0; t < 2; ++t) {
      v16bf kf[4];
#pragma unroll
      for (int c = 0; c < 4; ++c)
        kf[c] = *(const v16bf*)(Kb + (size_t)(jBase + c * 16 + lr) * DMODEL +
                                h * DHEAD + t * 32 + lh * 16);
#pragma unroll
      for (int c = 0; c < 4; ++c)
        s[c] = __builtin_amdgcn_wmma_f32_16x16x32_bf16(false, qf[t], false, kf[c],
                                                       (short)0, s[c], false, false);
    }

    int jcol[4], bj[4];
#pragma unroll
    for (int c = 0; c < 4; ++c) {
      jcol[c] = jBase + c * 16 + lr;
      bj[c] = bid[jcol[c]];
    }

    // mask + online softmax; rows r live entirely in this lane-half
#pragma unroll
    for (int r = 0; r < 8; ++r) {
      const int row = iBlk * 16 + lh * 8 + r;
      float v[4];
#pragma unroll
      for (int c = 0; c < 4; ++c) {
        const bool ok = (jcol[c] == 0) || ((bj[c] == bi[r]) && (jcol[c] < row));
        v[c] = ok ? s[c][r] : -3.0e38f;   // Q pre-scaled by 1/8
      }
      float mx = fmaxf(fmaxf(v[0], v[1]), fmaxf(v[2], v[3]));
#pragma unroll
      for (int d = 1; d < 16; d <<= 1) mx = fmaxf(mx, __shfl_xor(mx, d, 32));
      const float mNew = fmaxf(rowM[r], mx);
      const float corr = __expf(rowM[r] - mNew);
      float p[4], ps = 0.f;
#pragma unroll
      for (int c = 0; c < 4; ++c) { p[c] = __expf(v[c] - mNew); ps += p[c]; }
#pragma unroll
      for (int d = 1; d < 16; d <<= 1) ps += __shfl_xor(ps, d, 32);
      rowS[r] = rowS[r] * corr + ps;
      rowM[r] = mNew;
#pragma unroll
      for (int t = 0; t < 4; ++t) oAcc[t][r] *= corr;
#pragma unroll
      for (int c = 0; c < 4; ++c)
        Plds[(lh * 8 + r) * 64 + c * 16 + lr] = (bf16_t)p[c];
    }
    __syncthreads();

    // Transpose P (C layout) into two A fragments via LDS
    v16bf pf[2];
#pragma unroll
    for (int u = 0; u < 2; ++u) {
      const bf16_t* pb = Plds + lr * 64 + u * 32 + lh * 16;
      v8bf p0v = *(const v8bf*)(pb);
      v8bf p1v = *(const v8bf*)(pb + 8);
#pragma unroll
      for (int i = 0; i < 8; ++i) { pf[u][i] = p0v[i]; pf[u][i + 8] = p1v[i]; }
    }
    __syncthreads();

    // O += P @ V  (B fragments K-contiguous from V^T); batch loads per u
#pragma unroll
    for (int u = 0; u < 2; ++u) {
      v16bf vf[4];
#pragma unroll
      for (int t = 0; t < 4; ++t)
        vf[t] = *(const v16bf*)(Vt + (size_t)(h * DHEAD + t * 16 + lr) * LSEQ +
                                jBase + u * 32 + lh * 16);
#pragma unroll
      for (int t = 0; t < 4; ++t)
        oAcc[t] = __builtin_amdgcn_wmma_f32_16x16x32_bf16(false, pf[u], false, vf[t],
                                                          (short)0, oAcc[t], false, false);
    }
  }

  // normalize + store bf16
#pragma unroll
  for (int t = 0; t < 4; ++t) {
    const int col = h * DHEAD + t * 16 + lr;
#pragma unroll
    for (int r = 0; r < 8; ++r) {
      const int row = iBlk * 16 + lh * 8 + r;
      O[(size_t)row * DMODEL + col] = (bf16_t)(oAcc[t][r] / rowS[r]);
    }
  }
}

// ---------------------------------------------------------------------------
// LayerNorm over D=512, optional residual add: out = LN(x (+ res)) * g + b
// ---------------------------------------------------------------------------
__global__ __launch_bounds__(256)
void layernorm_kernel(const float* __restrict__ x, const float* __restrict__ res,
                      const float* __restrict__ g, const float* __restrict__ b,
                      float* __restrict__ outF, bf16_t* __restrict__ outB) {
  const int tok = blockIdx.x;
  const int tid = threadIdx.x;
  const size_t base = (size_t)tok * DMODEL;
  float v0 = x[base + tid];
  float v1 = x[base + tid + 256];
  if (res) { v0 += res[base + tid]; v1 += res[base + tid + 256]; }

  __shared__ float shs[256], shq[256];
  shs[tid] = v0 + v1;
  shq[tid] = v0 * v0 + v1 * v1;
  __syncthreads();
  for (int off = 128; off > 0; off >>= 1) {
    if (tid < off) { shs[tid] += shs[tid + off]; shq[tid] += shq[tid + off]; }
    __syncthreads();
  }
  const float mean = shs[0] * (1.f / DMODEL);
  const float var  = shq[0] * (1.f / DMODEL) - mean * mean;
  const float rstd = rsqrtf(var + LN_EPS);

  const float o0 = (v0 - mean) * rstd * g[tid] + b[tid];
  const float o1 = (v1 - mean) * rstd * g[tid + 256] + b[tid + 256];
  if (outF) { outF[base + tid] = o0; outF[base + tid + 256] = o1; }
  if (outB) { outB[base + tid] = (bf16_t)o0; outB[base + tid + 256] = (bf16_t)o1; }
}

// ---------------------------------------------------------------------------
// Host-side orchestration
// ---------------------------------------------------------------------------
extern "C" void kernel_launch(void* const* d_in, const int* in_sizes, int n_in,
                              void* d_out, int out_size, void* d_ws, size_t ws_size,
                              hipStream_t stream) {
  (void)in_sizes; (void)n_in; (void)out_size; (void)ws_size;

  const int*   hist = (const int*)d_in[0];
  const int*   sm   = (const int*)d_in[1];
  const float* Wt = (const float*)d_in[2];  const float* bt = (const float*)d_in[3];
  const float* Wa = (const float*)d_in[4];  const float* ba = (const float*)d_in[5];
  const float* Wm = (const float*)d_in[6];  const float* bm = (const float*)d_in[7];
  const float* sos = (const float*)d_in[8];
  const float* Wq = (const float*)d_in[9];  const float* bq = (const float*)d_in[10];
  const float* Wk = (const float*)d_in[11]; const float* bk = (const float*)d_in[12];
  const float* Wv = (const float*)d_in[13]; const float* bv = (const float*)d_in[14];
  const float* Wo = (const float*)d_in[15]; const float* bo = (const float*)d_in[16];
  const float* W1 = (const float*)d_in[17]; const float* b1 = (const float*)d_in[18];
  const float* W2 = (const float*)d_in[19]; const float* b2 = (const float*)d_in[20];
  const float* g1 = (const float*)d_in[21]; const float* be1 = (const float*)d_in[22];
  const float* g2 = (const float*)d_in[23]; const float* be2 = (const float*)d_in[24];

  // workspace carving (256B aligned)
  char* wp = (char*)d_ws;
  auto carve = [&](size_t bytes) -> void* {
    void* p = (void*)wp;
    wp += (bytes + 255) & ~(size_t)255;
    return p;
  };
  const size_t LD2 = (size_t)LSEQ * DMODEL * sizeof(bf16_t);
  const size_t LD4 = (size_t)LSEQ * DMODEL * sizeof(float);
  const size_t WW2 = (size_t)DMODEL * DMODEL * sizeof(bf16_t);

  int*    bidb = (int*)carve(LSEQ * sizeof(int));
  bf16_t* xb   = (bf16_t*)carve(LD2);
  bf16_t* qb   = (bf16_t*)carve(LD2);
  bf16_t* kb   = (bf16_t*)carve(LD2);
  bf16_t* vt   = (bf16_t*)carve(LD2);     // V^T : [D][L]
  bf16_t* aob  = (bf16_t*)carve(LD2);     // attention output (bf16)
  float*  aF   = (float*)carve(LD4);      // Wo output (pre-LN1)
  float*  xaF  = (float*)carve(LD4);      // LN1 output
  bf16_t* xab  = (bf16_t*)carve(LD2);
  bf16_t* h1b  = (bf16_t*)carve((size_t)LSEQ * FFDIM * sizeof(bf16_t));
  float*  yF   = (float*)carve(LD4);      // FF output (pre-LN2)
  bf16_t* wqb  = (bf16_t*)carve(WW2);
  bf16_t* wkb  = (bf16_t*)carve(WW2);
  bf16_t* wvb  = (bf16_t*)carve(WW2);
  bf16_t* wob  = (bf16_t*)carve(WW2);
  bf16_t* w1b  = (bf16_t*)carve(WW2);
  bf16_t* w2b  = (bf16_t*)carve(WW2);

  // weight conversion (deterministic, every launch)
  const int NW = DMODEL * DMODEL;
  f32_to_bf16_kernel<<<512, 256, 0, stream>>>(Wq, wqb, NW);
  f32_to_bf16_kernel<<<512, 256, 0, stream>>>(Wk, wkb, NW);
  f32_to_bf16_kernel<<<512, 256, 0, stream>>>(Wv, wvb, NW);
  f32_to_bf16_kernel<<<512, 256, 0, stream>>>(Wo, wob, NW);
  f32_to_bf16_kernel<<<512, 256, 0, stream>>>(W1, w1b, NW);
  f32_to_bf16_kernel<<<512, 256, 0, stream>>>(W2, w2b, NW);

  bid_kernel<<<1, 32, 0, stream>>>(sm, bidb);
  embed_kernel<<<LSEQ, 256, 0, stream>>>(hist, Wt, bt, Wa, ba, Wm, bm, sos, xb);

  const dim3 gGemm(LSEQ / 64, DMODEL / 64);   // (48, 8)
  const dim3 gAttn(LSEQ / 16, NHEAD);         // (192, 8)

  for (int layer = 0; layer < 2; ++layer) {
    // Q (pre-scaled by 1/sqrt(dh)), K, V projections (V stored transposed)
    gemm_bf16_kernel<<<gGemm, 128, 0, stream>>>(xb, wqb, bq, nullptr, qb, nullptr,
                                                LSEQ, DMODEL, DMODEL, 0, 0.125f);
    gemm_bf16_kernel<<<gGemm, 128, 0, stream>>>(xb, wkb, bk, nullptr, kb, nullptr,
                                                LSEQ, DMODEL, DMODEL, 0, 1.0f);
    gemm_bf16_kernel<<<gGemm, 128, 0, stream>>>(xb, wvb, bv, nullptr, nullptr, vt,
                                                LSEQ, DMODEL, DMODEL, 0, 1.0f);
    // masked attention
    attn_kernel<<<gAttn, 32, 0, stream>>>(qb, kb, vt, bidb, aob);
    // output projection (no attn residual in this model)
    gemm_bf16_kernel<<<gGemm, 128, 0, stream>>>(aob, wob, bo, aF, nullptr, nullptr,
                                                LSEQ, DMODEL, DMODEL, 0, 1.0f);
    layernorm_kernel<<<LSEQ, 256, 0, stream>>>(aF, nullptr, g1, be1, xaF, xab);
    // FF: leaky_relu(xa @ W1^T + b1) @ W2^T + b2
    gemm_bf16_kernel<<<gGemm, 128, 0, stream>>>(xab, w1b, b1, nullptr, h1b, nullptr,
                                                LSEQ, FFDIM, DMODEL, 1, 1.0f);
    gemm_bf16_kernel<<<gGemm, 128, 0, stream>>>(h1b, w2b, b2, yF, nullptr, nullptr,
                                                LSEQ, DMODEL, FFDIM, 0, 1.0f);
    // LN(xa + y): layer0 -> bf16 only (next layer input); layer1 -> d_out f32
    if (layer == 0)
      layernorm_kernel<<<LSEQ, 256, 0, stream>>>(yF, xaF, g2, be2, nullptr, xb);
    else
      layernorm_kernel<<<LSEQ, 256, 0, stream>>>(yF, xaF, g2, be2, (float*)d_out, nullptr);
  }
}
